// MetaGraphFCEncoderLayer_15401752724191
// MI455X (gfx1250) — matrix operand order repaired and checked by reading, not compiled, not run on hardware
//
#include <hip/hip_runtime.h>
#include <hip/hip_bf16.h>

// ---------------------------------------------------------------------------
// MetaGraphFCEncoderLayer on gfx1250 (MI455X, wave32, WMMA)
// Split-bf16 (hi+lo) GEMMs via v_wmma_f32_16x16x32_bf16.
// ---------------------------------------------------------------------------

typedef __attribute__((ext_vector_type(16))) __bf16 bf16x16_t;
typedef __attribute__((ext_vector_type(8)))  __bf16 bf16x8_t;
typedef __attribute__((ext_vector_type(8)))  float  floatx8_t;

static __device__ __forceinline__ void f32_to_hilo(float x, __bf16& h, __bf16& l) {
    h = (__bf16)x;                 // RNE truncation to bf16 (8 mantissa bits)
    l = (__bf16)(x - (float)h);    // residual (next ~8 bits)
}

union PackBF4 { __bf16 b[4]; unsigned long long u64; };

// ---------------------------------------------------------------------------
// Kernel 0: prep — bf16 hi/lo weight copies into workspace; pass-through outs.
//   nodeW [128][128] and edgeW [128][256] are already [N][K] row-major,
//   which is exactly the B-operand feed order for WMMA.
// ---------------------------------------------------------------------------
__global__ __launch_bounds__(256)
void prep_kernel(const float* __restrict__ nodeW, const float* __restrict__ edgeW,
                 const int* __restrict__ eidx, const float* __restrict__ u,
                 __bf16* __restrict__ nWhi, __bf16* __restrict__ nWlo,
                 __bf16* __restrict__ eWhi, __bf16* __restrict__ eWlo,
                 float* __restrict__ idx_out, float* __restrict__ u_out, int E) {
    const int i = blockIdx.x * blockDim.x + threadIdx.x;
    const int NW = 128 * 128;
    const int EW = 128 * 256;
    if (i < NW) { __bf16 h, l; f32_to_hilo(nodeW[i], h, l); nWhi[i] = h; nWlo[i] = l; }
    if (i < EW) { __bf16 h, l; f32_to_hilo(edgeW[i], h, l); eWhi[i] = h; eWlo[i] = l; }
    if (i < 2 * E) idx_out[i] = (float)eidx[i];   // indices < 2^24: exact in f32
    if (i < 128)   u_out[i] = u[i];
}

// ---------------------------------------------------------------------------
// Shared WMMA micro-kernel pieces
//   A (16x32 bf16) lane layout: lanes 0-15 -> M=lane, K = kb..kb+15
//                               lanes 16-31 -> M=lane-16, kb += 16
//   B (32x16 bf16) lane layout: N = lane%16, K = 32c + 16*(lane/16) .. +15
//   C (16x16 f32):  VGPR r -> M = r + 8*(lane/16), N = lane%16
// ---------------------------------------------------------------------------
static __device__ __forceinline__ bf16x16_t load_a_chunk(const __bf16* __restrict__ row, int kb) {
    bf16x8_t t0 = *(const bf16x8_t*)(row + kb);
    bf16x8_t t1 = *(const bf16x8_t*)(row + kb + 8);
    bf16x16_t a;
#pragma unroll
    for (int i = 0; i < 8; ++i) { a[i] = t0[i]; a[i + 8] = t1[i]; }
    return a;
}

// ---------------------------------------------------------------------------
// Kernel 1: node projection  [N,128] @ W[128,128]^T + b
// ---------------------------------------------------------------------------
#define NODE_ROWS 128
#define KA_N (128 + 8)   // padded LDS row stride (bf16 elems) -> 272B, 16B aligned

__global__ __launch_bounds__(256)
void node_gemm_kernel(const float* __restrict__ xfeat,
                      const __bf16* __restrict__ Whi, const __bf16* __restrict__ Wlo,
                      const float* __restrict__ bias, float* __restrict__ out, int N) {
    __shared__ __bf16 Ahi[NODE_ROWS * KA_N];
    __shared__ __bf16 Alo[NODE_ROWS * KA_N];

    const int tid  = threadIdx.x;
    const int base = blockIdx.x * NODE_ROWS;

    // Stage A tile: 128 rows x 128 feats, f32 -> bf16 hi/lo in LDS
    for (int j = tid; j < NODE_ROWS * 32; j += 256) {
        const int r = j >> 5, k0 = (j & 31) * 4;
        const int row = base + r;
        float4 v = make_float4(0.f, 0.f, 0.f, 0.f);
        if (row < N) v = *(const float4*)(xfeat + (size_t)row * 128 + k0);
        PackBF4 ph, pl;
        f32_to_hilo(v.x, ph.b[0], pl.b[0]); f32_to_hilo(v.y, ph.b[1], pl.b[1]);
        f32_to_hilo(v.z, ph.b[2], pl.b[2]); f32_to_hilo(v.w, ph.b[3], pl.b[3]);
        *(unsigned long long*)&Ahi[r * KA_N + k0] = ph.u64;
        *(unsigned long long*)&Alo[r * KA_N + k0] = pl.u64;
    }
    __syncthreads();

    const int lane  = tid & 31;
    const int wv    = tid >> 5;             // wave id = N tile (8 waves x 16 cols)
    const int khalf = lane >> 4;
    const int ncol  = wv * 16 + (lane & 15);

    // B slice (this wave's 16 columns, full K=128, hi+lo) lives in VGPRs
    bf16x16_t Bh[4], Bl[4];
#pragma unroll
    for (int c = 0; c < 4; ++c) {
        const size_t off = (size_t)ncol * 128 + c * 32 + khalf * 16;
        Bh[c] = *(const bf16x16_t*)(Whi + off);
        Bl[c] = *(const bf16x16_t*)(Wlo + off);
    }
    const float bv = bias[ncol];

#pragma unroll
    for (int m = 0; m < 8; ++m) {
        const __bf16* arow_h = Ahi + (size_t)(m * 16 + (lane & 15)) * KA_N;
        const __bf16* arow_l = Alo + (size_t)(m * 16 + (lane & 15)) * KA_N;
        floatx8_t acc = {};
#pragma unroll
        for (int c = 0; c < 4; ++c) {
            const int kb = c * 32 + khalf * 16;
            bf16x16_t ah = load_a_chunk(arow_h, kb);
            bf16x16_t al = load_a_chunk(arow_l, kb);
            acc = __builtin_amdgcn_wmma_f32_16x16x32_bf16(false, ah, false, Bh[c], (short)0, acc, false, false);
            acc = __builtin_amdgcn_wmma_f32_16x16x32_bf16(false, ah, false, Bl[c], (short)0, acc, false, false);
            acc = __builtin_amdgcn_wmma_f32_16x16x32_bf16(false, al, false, Bh[c], (short)0, acc, false, false);
        }
        const long rbase = (long)base + m * 16 + 8 * khalf;
#pragma unroll
        for (int r = 0; r < 8; ++r) {
            const long row = rbase + r;
            if (row < N) out[row * 128 + ncol] = acc[r] + bv;
        }
    }
}

// ---------------------------------------------------------------------------
// Kernel 2: edge MLP  concat(x[src], x[dst]) @ edgeW[128,256]^T + b
// Gather hits L2 (51MB feature table << 192MB L2).
// ---------------------------------------------------------------------------
#define EDGE_ROWS 64
#define KA_E (256 + 8)   // padded LDS row stride -> 528B, 16B aligned

__global__ __launch_bounds__(256)
void edge_gemm_kernel(const float* __restrict__ xfeat, const int* __restrict__ eidx,
                      const __bf16* __restrict__ Whi, const __bf16* __restrict__ Wlo,
                      const float* __restrict__ bias, float* __restrict__ out, int E) {
    __shared__ __bf16 Ahi[EDGE_ROWS * KA_E];
    __shared__ __bf16 Alo[EDGE_ROWS * KA_E];
    __shared__ int    sIdx[2 * EDGE_ROWS];

    const int tid  = threadIdx.x;
    const int base = blockIdx.x * EDGE_ROWS;

    if (tid < 2 * EDGE_ROWS) {
        const int which = tid >> 6;   // 0 = src endpoints, 1 = dst endpoints
        const int e     = tid & 63;
        const int ge    = base + e;
        sIdx[tid] = (ge < E) ? eidx[(size_t)which * E + ge] : 0;
    }
    __syncthreads();

    // Stage gathered A tile: 64 edges x 256 (src||dst) feats as bf16 hi/lo
    for (int j = tid; j < EDGE_ROWS * 64; j += 256) {
        const int e  = j >> 6;
        const int k0 = (j & 63) * 4;           // 0..252
        const int which = (k0 >= 128) ? 1 : 0;
        const int kf = k0 - which * 128;
        const int row = sIdx[which * 64 + e];
        const float4 v = *(const float4*)(xfeat + (size_t)row * 128 + kf);
        PackBF4 ph, pl;
        f32_to_hilo(v.x, ph.b[0], pl.b[0]); f32_to_hilo(v.y, ph.b[1], pl.b[1]);
        f32_to_hilo(v.z, ph.b[2], pl.b[2]); f32_to_hilo(v.w, ph.b[3], pl.b[3]);
        *(unsigned long long*)&Ahi[e * KA_E + k0] = ph.u64;
        *(unsigned long long*)&Alo[e * KA_E + k0] = pl.u64;
    }
    __syncthreads();

    const int lane  = tid & 31;
    const int wv    = tid >> 5;
    const int khalf = lane >> 4;
    const int ncol  = wv * 16 + (lane & 15);

    // Full-K B slice in VGPRs: 8 chunks x (hi+lo) x 8 VGPRs = 128 VGPRs
    bf16x16_t Bh[8], Bl[8];
#pragma unroll
    for (int c = 0; c < 8; ++c) {
        const size_t off = (size_t)ncol * 256 + c * 32 + khalf * 16;
        Bh[c] = *(const bf16x16_t*)(Whi + off);
        Bl[c] = *(const bf16x16_t*)(Wlo + off);
    }
    const float bv = bias[ncol];

#pragma unroll
    for (int m = 0; m < 4; ++m) {
        const __bf16* arow_h = Ahi + (size_t)(m * 16 + (lane & 15)) * KA_E;
        const __bf16* arow_l = Alo + (size_t)(m * 16 + (lane & 15)) * KA_E;
        floatx8_t acc = {};
#pragma unroll
        for (int c = 0; c < 8; ++c) {
            const int kb = c * 32 + khalf * 16;
            bf16x16_t ah = load_a_chunk(arow_h, kb);
            bf16x16_t al = load_a_chunk(arow_l, kb);
            acc = __builtin_amdgcn_wmma_f32_16x16x32_bf16(false, ah, false, Bh[c], (short)0, acc, false, false);
            acc = __builtin_amdgcn_wmma_f32_16x16x32_bf16(false, ah, false, Bl[c], (short)0, acc, false, false);
            acc = __builtin_amdgcn_wmma_f32_16x16x32_bf16(false, al, false, Bh[c], (short)0, acc, false, false);
        }
        const long rbase = (long)base + m * 16 + 8 * khalf;
#pragma unroll
        for (int r = 0; r < 8; ++r) {
            const long row = rbase + r;
            if (row < E) out[row * 128 + ncol] = acc[r] + bv;
        }
    }
}

// ---------------------------------------------------------------------------
// Launcher
// ---------------------------------------------------------------------------
extern "C" void kernel_launch(void* const* d_in, const int* in_sizes, int n_in,
                              void* d_out, int out_size, void* d_ws, size_t ws_size,
                              hipStream_t stream) {
    const float* node_features = (const float*)d_in[0];
    const int*   edge_indexes  = (const int*)  d_in[1];
    // d_in[2] = edge_features (ignored, hasEdgeInfo=False)
    const float* u_features    = (const float*)d_in[3];
    const float* node_W        = (const float*)d_in[4];
    const float* node_b        = (const float*)d_in[5];
    const float* edge_W        = (const float*)d_in[6];
    const float* edge_b        = (const float*)d_in[7];

    const int FEAT = 128, CH = 128;
    const int N = in_sizes[0] / FEAT;   // 100000
    const int E = in_sizes[1] / 2;      // 500000

    float* out      = (float*)d_out;
    float* node_out = out;                                 // [N, CH]
    float* idx_out  = node_out + (size_t)N * CH;           // [2, E] as float
    float* edge_out = idx_out + (size_t)2 * E;             // [E, CH]
    float* u_out    = edge_out + (size_t)E * CH;           // [1, CH]

    // Workspace: bf16 hi/lo weight copies (192 KB total)
    __bf16* nWhi = (__bf16*)d_ws;
    __bf16* nWlo = nWhi + 128 * 128;
    __bf16* eWhi = nWlo + 128 * 128;
    __bf16* eWlo = eWhi + 128 * 256;

    int prepN = 2 * E;
    if (prepN < 128 * 256) prepN = 128 * 256;
    prep_kernel<<<(prepN + 255) / 256, 256, 0, stream>>>(
        node_W, edge_W, edge_indexes, u_features,
        nWhi, nWlo, eWhi, eWlo, idx_out, u_out, E);

    node_gemm_kernel<<<(N + NODE_ROWS - 1) / NODE_ROWS, 256, 0, stream>>>(
        node_features, nWhi, nWlo, node_b, node_out, N);

    edge_gemm_kernel<<<(E + EDGE_ROWS - 1) / EDGE_ROWS, 256, 0, stream>>>(
        node_features, edge_indexes, eWhi, eWlo, edge_b, edge_out, E);
}
